// InterpretableMultimodalCapsuleFusion_78340203479693
// MI455X (gfx1250) — matrix operand order repaired and checked by reading, not compiled, not run on hardware
//
#include <hip/hip_runtime.h>
#include <hip/hip_bf16.h>
#include <math.h>

// ---------------------------------------------------------------------------
// InterpretableMultimodalCapsuleFusion for MI455X (gfx1250, wave32, WMMA).
//
// - All GEMM-shaped work goes through v_wmma_f32_16x16x32_f16 (f32 accum).
// - Big input-projection GEMMs take a pre-converted, K-padded f16 A operand
//   and stage A tiles with GLOBAL_LOAD_ASYNC_TO_LDS_B128 (+ s_wait_asynccnt).
// - LSTM recurrence: one 128-thread block owns a private 16-row batch tile;
//   h/c/gates live in LDS, full time loop runs without inter-block sync.
//   H=64 keeps Whh resident in LDS as f16.
// ---------------------------------------------------------------------------

typedef _Float16 f16;
typedef __attribute__((ext_vector_type(16))) _Float16 v16h;
typedef __attribute__((ext_vector_type(8)))  float    v8f;

#define BSZ 1024
#define TSEQ 128

__device__ __forceinline__ float sigm(float x) { return 1.0f / (1.0f + __expf(-x)); }

// WMMA lane layouts used here (A/B in this file's contiguous k order, not the ISA table order):
//  A (16x32 f16): lane = m + 16*(k>=16),     elem = k%16
//  B (32x16 f16): lane = n + 16*(k>=16),     elem = k%16
//  C (16x16 f32): lane = n + 16*(m>=8),      vgpr = m%8
__device__ __forceinline__ int a_kmap(int e, bool hi) {
    return e + (hi ? 16 : 0);
}

// ---------------------------------------------------------------------------
// Pad + convert: X (M x K, f32) -> Xh (M x KP, f16), zero-padded K -> KP.
// ---------------------------------------------------------------------------
__global__ void pad_convert_kernel(const float* __restrict__ X, f16* __restrict__ Xh,
                                   int K, int KP, long M) {
    long i = (long)blockIdx.x * blockDim.x + threadIdx.x;
    if (i >= M * (long)KP) return;
    long r = i / KP;
    int  c = (int)(i - r * KP);
    Xh[i] = (c < K) ? (f16)X[r * (long)K + c] : (f16)0.0f;
}

// ---------------------------------------------------------------------------
// Generic GEMM:  C = A @ op(W) + bias
//   AF16=true : A is f16, (M x lda) with lda = K padded to 32 (no A bounds);
//               A tiles staged via async global->LDS copies.
//   AF16=false: A is f32, K multiple of 32, lda multiple of 4.
//   WT=true : W is (N,KW) row-major -> C = A @ W^T
//   WT=false: W is (KW,N) row-major -> C = A @ W
// Block: 128 threads = 4 waves, 64x64 output tile, K staged in 32-chunks.
// ---------------------------------------------------------------------------
template <bool WT, bool OUTF16, bool AF16>
__global__ void wmma_gemm_kernel(const void* __restrict__ Ap, int lda,
                                 const float* __restrict__ W,
                                 const float* __restrict__ bias,
                                 void* __restrict__ Cv, int ldc,
                                 int K /*loop bound*/, int KW /*real K for W*/,
                                 int N) {
    __shared__ __align__(16) f16 Asub[64][32];
    __shared__ __align__(16) f16 Wsub[64][32];   // indexed [n][k]

    const int tid  = threadIdx.x;
    const int wv   = tid >> 5;
    const int lane = tid & 31;
    const int ln   = lane & 15;
    const bool hi  = lane >= 16;
    const int mBase = blockIdx.x * 64;
    const int nBase = blockIdx.y * 64;

    v8f acc[4];
#pragma unroll
    for (int nt = 0; nt < 4; ++nt) {
        float bv = bias ? bias[nBase + nt * 16 + ln] : 0.0f;
#pragma unroll
        for (int r = 0; r < 8; ++r) acc[nt][r] = bv;
    }

    for (int kk = 0; kk < K; kk += 32) {
        // ---- stage A tile ----
        if (AF16) {
            const f16* Ah = (const f16*)Ap;
            // 64 rows x 64B; 256 x 16B chunks; async DMA into LDS
#pragma unroll
            for (int it = 0; it < 2; ++it) {
                int u = tid + it * 128;
                int r = u >> 2, c = (u & 3) * 8;
                const f16* gp = Ah + (size_t)(mBase + r) * lda + kk + c;
                unsigned lo = (unsigned)(uintptr_t)(&Asub[r][c]);
                asm volatile("global_load_async_to_lds_b128 %0, %1, off"
                             :: "v"(lo), "v"(gp) : "memory");
            }
        } else {
            const float* Af = (const float*)Ap;
            if ((lda & 3) == 0) {
                for (int u = tid; u < 64 * 8; u += 128) {
                    int r = u >> 3, c = (u & 7) * 4;
                    float4 a4 = *(const float4*)(Af + (size_t)(mBase + r) * lda + kk + c);
                    Asub[r][c + 0] = (f16)a4.x; Asub[r][c + 1] = (f16)a4.y;
                    Asub[r][c + 2] = (f16)a4.z; Asub[r][c + 3] = (f16)a4.w;
                }
            } else {
                for (int i = tid; i < 64 * 32; i += 128) {
                    int r = i >> 5, c = i & 31;
                    Asub[r][c] = (f16)Af[(size_t)(mBase + r) * lda + kk + c];
                }
            }
        }

        // ---- stage W tile (f32 -> f16) ----
        if (kk + 32 <= KW) {               // uniform fast path, no lane guards
            if (WT && ((KW & 3) == 0)) {
                for (int u = tid; u < 64 * 8; u += 128) {
                    int r = u >> 3, c = (u & 7) * 4;
                    float4 w4 = *(const float4*)(W + (size_t)(nBase + r) * KW + kk + c);
                    Wsub[r][c + 0] = (f16)w4.x; Wsub[r][c + 1] = (f16)w4.y;
                    Wsub[r][c + 2] = (f16)w4.z; Wsub[r][c + 3] = (f16)w4.w;
                }
            } else {
                for (int i = tid; i < 64 * 32; i += 128) {
                    int r = i >> 5, c = i & 31;
                    int gk = kk + c;
                    float wval = WT ? W[(size_t)(nBase + r) * KW + gk]
                                    : W[(size_t)gk * N + (nBase + r)];
                    Wsub[r][c] = (f16)wval;
                }
            }
        } else {                           // K edge: guarded, zero-filled
            for (int i = tid; i < 64 * 32; i += 128) {
                int r = i >> 5, c = i & 31;
                int gk = kk + c;
                float wval = 0.0f;
                if (gk < KW)
                    wval = WT ? W[(size_t)(nBase + r) * KW + gk]
                              : W[(size_t)gk * N + (nBase + r)];
                Wsub[r][c] = (f16)wval;
            }
        }

        if (AF16) asm volatile("s_wait_asynccnt 0" ::: "memory");
        __syncthreads();

        v16h afr;
#pragma unroll
        for (int e = 0; e < 16; ++e) afr[e] = Asub[wv * 16 + ln][a_kmap(e, hi)];

#pragma unroll
        for (int nt = 0; nt < 4; ++nt) {
            v16h bfr;
#pragma unroll
            for (int e = 0; e < 16; ++e) bfr[e] = Wsub[nt * 16 + ln][e + (hi ? 16 : 0)];
            acc[nt] = __builtin_amdgcn_wmma_f32_16x16x32_f16(
                false, afr, false, bfr, (short)0, acc[nt], false, false);
        }
        __syncthreads();
    }

#pragma unroll
    for (int nt = 0; nt < 4; ++nt) {
        int col = nBase + nt * 16 + ln;
#pragma unroll
        for (int r = 0; r < 8; ++r) {
            int row = mBase + wv * 16 + r + (hi ? 8 : 0);
            if (OUTF16) ((f16*)Cv)[(size_t)row * ldc + col] = (f16)acc[nt][r];
            else        ((float*)Cv)[(size_t)row * ldc + col] = acc[nt][r];
        }
    }
}

// ---------------------------------------------------------------------------
// LSTM over precomputed input projections.
//   proj : (B*Tlen, 4*HH) f16, already contains x@Wih.T + b, row = b*Tlen + t
//   Whh  : (4*HH, HH) f32
//   h_out: final hidden, h_out[b*stride + c]
// One block = 16 batch rows, 4 waves; wave w owns gate cols [w*HH, (w+1)*HH).
// HH==64: Whh preloaded into LDS as f16. HH==128: float4 global reads.
// ---------------------------------------------------------------------------
template <int HH>
__global__ void lstm_kernel(const f16* __restrict__ proj,
                            const float* __restrict__ Whh,
                            float* __restrict__ h_out, int h_out_stride,
                            int Tlen, int rev) {
    constexpr int G  = 4 * HH;
    constexpr int NT = HH / 16;

    __shared__ __align__(16) f16 hbf[16][HH];
    __shared__ float cbuf[16][HH];
    __shared__ float hfl[16][HH];
    __shared__ float gates[16][G];
    __shared__ __align__(16) f16 wlds[(HH == 64) ? (G * HH) : 1];

    const int tid  = threadIdx.x;
    const int wv   = tid >> 5;
    const int lane = tid & 31;
    const int ln   = lane & 15;
    const bool hi  = lane >= 16;
    const int bBase = blockIdx.x * 16;

    if constexpr (HH == 64) {
        for (int u = tid; u < (G * HH) / 4; u += 128) {
            int idx = u * 4;
            float4 w4 = *(const float4*)(Whh + idx);
            wlds[idx + 0] = (f16)w4.x; wlds[idx + 1] = (f16)w4.y;
            wlds[idx + 2] = (f16)w4.z; wlds[idx + 3] = (f16)w4.w;
        }
    }
    for (int i = tid; i < 16 * HH; i += 128) {
        int r = i / HH, c = i % HH;
        hbf[r][c]  = (f16)0.0f;
        cbuf[r][c] = 0.0f;
    }
    __syncthreads();

    for (int s = 0; s < Tlen; ++s) {
        int t = rev ? (Tlen - 1 - s) : s;

        v8f acc[NT];
#pragma unroll
        for (int nt = 0; nt < NT; ++nt) {
            int col = wv * HH + nt * 16 + ln;
#pragma unroll
            for (int r = 0; r < 8; ++r) {
                int row = r + (hi ? 8 : 0);
                acc[nt][r] = (float)proj[((size_t)(bBase + row) * Tlen + t) * G + col];
            }
        }

        for (int kk = 0; kk < HH; kk += 32) {
            v16h afr;
#pragma unroll
            for (int e = 0; e < 16; ++e) afr[e] = hbf[ln][kk + a_kmap(e, hi)];
#pragma unroll
            for (int nt = 0; nt < NT; ++nt) {
                int ncol = wv * HH + nt * 16 + ln;
                v16h bfr;
                if constexpr (HH == 64) {
#pragma unroll
                    for (int e = 0; e < 16; ++e)
                        bfr[e] = wlds[ncol * HH + kk + e + (hi ? 16 : 0)];
                } else {
#pragma unroll
                    for (int e4 = 0; e4 < 4; ++e4) {
                        float4 w4 = *(const float4*)(
                            Whh + (size_t)ncol * HH + kk + (hi ? 16 : 0) + e4 * 4);
                        bfr[e4 * 4 + 0] = (f16)w4.x; bfr[e4 * 4 + 1] = (f16)w4.y;
                        bfr[e4 * 4 + 2] = (f16)w4.z; bfr[e4 * 4 + 3] = (f16)w4.w;
                    }
                }
                acc[nt] = __builtin_amdgcn_wmma_f32_16x16x32_f16(
                    false, afr, false, bfr, (short)0, acc[nt], false, false);
            }
        }

#pragma unroll
        for (int nt = 0; nt < NT; ++nt) {
            int col = wv * HH + nt * 16 + ln;
#pragma unroll
            for (int r = 0; r < 8; ++r) gates[r + (hi ? 8 : 0)][col] = acc[nt][r];
        }
        __syncthreads();

        for (int i = tid; i < 16 * HH; i += 128) {
            int r = i / HH, c = i % HH;
            float gi = gates[r][c];
            float gf = gates[r][HH + c];
            float gg = gates[r][2 * HH + c];
            float go = gates[r][3 * HH + c];
            float cc = sigm(gf) * cbuf[r][c] + sigm(gi) * tanhf(gg);
            cbuf[r][c] = cc;
            float hh = sigm(go) * tanhf(cc);
            hfl[r][c] = hh;
            hbf[r][c] = (f16)hh;
        }
        __syncthreads();
    }

    for (int i = tid; i < 16 * HH; i += 128) {
        int r = i / HH, c = i % HH;
        h_out[(size_t)(bBase + r) * h_out_stride + c] = hfl[r][c];
    }
}

// ---------------------------------------------------------------------------
// Small elementwise / reduction kernels
// ---------------------------------------------------------------------------
__global__ void fill_kernel(float* p, float v, int n) {
    int i = blockIdx.x * blockDim.x + threadIdx.x;
    if (i < n) p[i] = v;
}

__global__ void mul_kernel(const float* a, const float* b, float* o, int n) {
    int i = blockIdx.x * blockDim.x + threadIdx.x;
    if (i < n) o[i] = a[i] * b[i];
}

__global__ void add_kernel(const float* a, const float* b, float* o, int n) {
    int i = blockIdx.x * blockDim.x + threadIdx.x;
    if (i < n) o[i] = a[i] + b[i];
}

// softmax over capsule axis (n <= 7) of rc (B, n, 128); one thread per (b, d)
__global__ void softmax1_kernel(float* rc, int n) {
    int i = blockIdx.x * blockDim.x + threadIdx.x;
    if (i >= BSZ * 128) return;
    int b = i >> 7, d = i & 127;
    float* base = rc + (size_t)b * n * 128 + d;
    float mx = -1e30f;
    for (int j = 0; j < n; ++j) mx = fmaxf(mx, base[(size_t)j * 128]);
    float e[8], s = 0.0f;
    for (int j = 0; j < n; ++j) { e[j] = __expf(base[(size_t)j * 128] - mx); s += e[j]; }
    float inv = 1.0f / s;
    for (int j = 0; j < n; ++j) base[(size_t)j * 128] = e[j] * inv;
}

// rc[b,j,:] += dot(pre[b,j,:], bc[b,:]); one thread per (b, j)
__global__ void routing_update_kernel(float* rc, const float* pre,
                                      const float* bc, int bc_stride, int n) {
    int i = blockIdx.x * blockDim.x + threadIdx.x;
    if (i >= BSZ * n) return;
    int b = i / n, j = i % n;
    const float* pr = pre + ((size_t)b * n + j) * 128;
    const float* bv = bc + (size_t)b * bc_stride;
    float d = 0.0f;
    for (int k = 0; k < 128; ++k) d += pr[k] * bv[k];
    float* rr = rc + ((size_t)b * n + j) * 128;
    for (int k = 0; k < 128; ++k) rr[k] += d;
}

// out[b] = fc2_b + sum_j tanh(o1pre[b,j]) * fc2_W[j]
__global__ void fc_head_kernel(const float* o1pre, const float* fc2W,
                               const float* fc2b, float* out) {
    int b = blockIdx.x * blockDim.x + threadIdx.x;
    if (b >= BSZ) return;
    float s = fc2b[0];
    for (int j = 0; j < 64; ++j) s += tanhf(o1pre[(size_t)b * 64 + j]) * fc2W[j];
    out[b] = s;
}

// ---------------------------------------------------------------------------
// Host orchestration
// ---------------------------------------------------------------------------
static inline void launch_gemm(const void* A, int lda, const float* W,
                               const float* bias, void* C, int ldc,
                               int M, int N, int K, int KW,
                               bool wt, bool outf16, bool af16, hipStream_t s) {
    dim3 g(M / 64, N / 64), b(128, 1, 1);
    if (af16) {
        // only WT+OUTF16 used with f16 A
        wmma_gemm_kernel<true, true, true><<<g, b, 0, s>>>(A, lda, W, bias, C, ldc, K, KW, N);
    } else if (wt) {
        if (outf16) wmma_gemm_kernel<true, true, false><<<g, b, 0, s>>>(A, lda, W, bias, C, ldc, K, KW, N);
        else        wmma_gemm_kernel<true, false, false><<<g, b, 0, s>>>(A, lda, W, bias, C, ldc, K, KW, N);
    } else {
        wmma_gemm_kernel<false, false, false><<<g, b, 0, s>>>(A, lda, W, bias, C, ldc, K, KW, N);
    }
}

extern "C" void kernel_launch(void* const* d_in, const int* in_sizes, int n_in,
                              void* d_out, int out_size, void* d_ws, size_t ws_size,
                              hipStream_t stream) {
    (void)in_sizes; (void)n_in; (void)out_size; (void)ws_size;
    const int B = BSZ, T = TSEQ;

    // -- inputs (setup_inputs dict order) --
    const float* text  = (const float*)d_in[0];
    const float* audio = (const float*)d_in[1];
    const float* video = (const float*)d_in[2];
    const float* Wt     = (const float*)d_in[21];
    const float* Wa     = (const float*)d_in[22];
    const float* Wv     = (const float*)d_in[23];
    const float* r_Wih  = (const float*)d_in[24];
    const float* r_Whh  = (const float*)d_in[25];
    const float* r_b    = (const float*)d_in[26];
    const float* dWih_f = (const float*)d_in[27];
    const float* dWhh_f = (const float*)d_in[28];
    const float* db_f   = (const float*)d_in[29];
    const float* dWih_b = (const float*)d_in[30];
    const float* dWhh_b = (const float*)d_in[31];
    const float* db_b   = (const float*)d_in[32];
    const float* fc1_W  = (const float*)d_in[33];
    const float* fc1_b  = (const float*)d_in[34];
    const float* fc2_W  = (const float*)d_in[35];
    const float* fc2_b  = (const float*)d_in[36];

    // -- workspace carve-up --
    size_t off = 0;
    auto alloc = [&](size_t bytes) -> void* {
        void* p = (char*)d_ws + off;
        off += (bytes + 255) & ~(size_t)255;
        return p;
    };
    f16*   Xh    = (f16*)alloc((size_t)B * T * 320 * sizeof(f16));   // padded f16 inputs
    f16*   projF = (f16*)alloc((size_t)B * T * 256 * sizeof(f16));   // reused @ N=512
    f16*   projB = (f16*)alloc((size_t)B * T * 256 * sizeof(f16));
    float* tc    = (float*)alloc((size_t)B * 128 * 4);
    float* ac    = (float*)alloc((size_t)B * 128 * 4);
    float* vc    = (float*)alloc((size_t)B * 128 * 4);
    float* pre0  = (float*)alloc((size_t)B * 2 * 128 * 4);
    float* pre1  = (float*)alloc((size_t)B * 2 * 128 * 4);
    float* pre2  = (float*)alloc((size_t)B * 2 * 128 * 4);
    float* pre3  = (float*)alloc((size_t)B * 3 * 128 * 4);
    float* deci  = (float*)alloc((size_t)B * 7 * 128 * 4);
    float* rc0   = (float*)alloc((size_t)B * 2 * 128 * 4);
    float* rc1   = (float*)alloc((size_t)B * 2 * 128 * 4);
    float* rc2   = (float*)alloc((size_t)B * 2 * 128 * 4);
    float* rc3   = (float*)alloc((size_t)B * 3 * 128 * 4);
    float* rc4   = (float*)alloc((size_t)B * 7 * 128 * 4);
    float* xprod = (float*)alloc((size_t)B * 7 * 128 * 4);
    float* dcf   = (float*)alloc((size_t)B * 128 * 4);
    float* dcb   = (float*)alloc((size_t)B * 128 * 4);
    float* dc    = (float*)alloc((size_t)B * 128 * 4);
    float* o1pre = (float*)alloc((size_t)B * 64 * 4);

    // ---- phase 1: modality BiLSTMs (H=64) ----
    struct Mod { const float* X; int din; int base; float* ctx; } mods[3] = {
        { text, 300, 3, tc }, { audio, 74, 9, ac }, { video, 35, 15, vc }
    };
    for (int m = 0; m < 3; ++m) {
        const float* WihF = (const float*)d_in[mods[m].base + 0];
        const float* WhhF = (const float*)d_in[mods[m].base + 1];
        const float* bF   = (const float*)d_in[mods[m].base + 2];
        const float* WihB = (const float*)d_in[mods[m].base + 3];
        const float* WhhB = (const float*)d_in[mods[m].base + 4];
        const float* bB   = (const float*)d_in[mods[m].base + 5];
        int din = mods[m].din;
        int KP  = (din + 31) & ~31;

        long cnt = (long)B * T * KP;
        pad_convert_kernel<<<(int)((cnt + 255) / 256), 256, 0, stream>>>(
            mods[m].X, Xh, din, KP, (long)B * T);

        launch_gemm(Xh, KP, WihF, bF, projF, 256, B * T, 256, KP, din, true, true, true, stream);
        launch_gemm(Xh, KP, WihB, bB, projB, 256, B * T, 256, KP, din, true, true, true, stream);
        lstm_kernel<64><<<B / 16, 128, 0, stream>>>(projF, WhhF, mods[m].ctx, 128, T, 0);
        lstm_kernel<64><<<B / 16, 128, 0, stream>>>(projB, WhhB, mods[m].ctx + 64, 128, T, 1);
    }

    // ---- phase 2: unimodal capsule transforms -> pre[] / deci slices ----
    struct Dst { float* p; int ldc; };
    Dst tD[4] = { { pre0 + 0,   256 }, { pre1 + 0,   256 }, { pre3 + 0,   384 }, { deci + 0,   896 } };
    Dst aD[4] = { { pre0 + 128, 256 }, { pre2 + 0,   256 }, { pre3 + 128, 384 }, { deci + 128, 896 } };
    Dst vD[4] = { { pre1 + 128, 256 }, { pre2 + 128, 256 }, { pre3 + 256, 384 }, { deci + 256, 896 } };
    for (int k = 0; k < 4; ++k) {
        launch_gemm(tc, 128, Wt + (size_t)k * 128 * 128, nullptr, tD[k].p, tD[k].ldc, B, 128, 128, 128, false, false, false, stream);
        launch_gemm(ac, 128, Wa + (size_t)k * 128 * 128, nullptr, aD[k].p, aD[k].ldc, B, 128, 128, 128, false, false, false, stream);
        launch_gemm(vc, 128, Wv + (size_t)k * 128 * 128, nullptr, vD[k].p, vD[k].ldc, B, 128, 128, 128, false, false, false, stream);
    }

    // ---- phase 3: dynamic routing ----
    float* rcs[5]  = { rc0, rc1, rc2, rc3, rc4 };
    float* pres[4] = { pre0, pre1, pre2, pre3 };
    int    ncap[5] = { 2, 2, 2, 3, 7 };
    for (int i = 0; i < 5; ++i) {
        int n = B * ncap[i] * 128;
        fill_kernel<<<(n + 255) / 256, 256, 0, stream>>>(rcs[i], 1.0f, n);
    }

    const int ROUTING = 3;
    for (int r = 0; r <= ROUTING; ++r) {
        for (int i = 0; i < 5; ++i)
            softmax1_kernel<<<(BSZ * 128 + 255) / 256, 256, 0, stream>>>(rcs[i], ncap[i]);

        // capsule LSTMs (H=128, forward only) -> bc[i] into deci slot 3+i
        for (int i = 0; i < 4; ++i) {
            int n = ncap[i];
            int cnt = B * n * 128;
            mul_kernel<<<(cnt + 255) / 256, 256, 0, stream>>>(rcs[i], pres[i], xprod, cnt);
            launch_gemm(xprod, 128, r_Wih + (size_t)i * 512 * 128, r_b + (size_t)i * 512,
                        projF, 512, B * n, 512, 128, 128, true, true, false, stream);
            lstm_kernel<128><<<B / 16, 128, 0, stream>>>(
                projF, r_Whh + (size_t)i * 512 * 128, deci + (3 + i) * 128, 896, n, 0);
        }

        // decision BiLSTM over (B, 7, 128)
        {
            int cnt = B * 7 * 128;
            mul_kernel<<<(cnt + 255) / 256, 256, 0, stream>>>(rc4, deci, xprod, cnt);
            launch_gemm(xprod, 128, dWih_f, db_f, projF, 512, B * 7, 512, 128, 128, true, true, false, stream);
            launch_gemm(xprod, 128, dWih_b, db_b, projB, 512, B * 7, 512, 128, 128, true, true, false, stream);
            lstm_kernel<128><<<B / 16, 128, 0, stream>>>(projF, dWhh_f, dcf, 128, 7, 0);
            lstm_kernel<128><<<B / 16, 128, 0, stream>>>(projB, dWhh_b, dcb, 128, 7, 1);
            add_kernel<<<(BSZ * 128 + 255) / 256, 256, 0, stream>>>(dcf, dcb, dc, B * 128);
        }

        if (r < ROUTING) {
            for (int i = 0; i < 4; ++i) {
                int thr = B * ncap[i];
                routing_update_kernel<<<(thr + 255) / 256, 256, 0, stream>>>(
                    rcs[i], pres[i], deci + (3 + i) * 128, 896, ncap[i]);
            }
            int thr = B * 7;
            routing_update_kernel<<<(thr + 255) / 256, 256, 0, stream>>>(
                rc4, deci, dc, 128, 7);
        }
    }

    // ---- phase 4: head ----
    launch_gemm(dc, 128, fc1_W, fc1_b, o1pre, 64, B, 64, 128, 128, true, false, false, stream);
    fc_head_kernel<<<(BSZ + 255) / 256, 256, 0, stream>>>(o1pre, fc2_W, fc2_b, (float*)d_out);
}
